// UniversalTransformerMultiLayerDecoder_62216896250436
// MI455X (gfx1250) — matrix operand order repaired and checked by reading, not compiled, not run on hardware
//
#include <hip/hip_runtime.h>
#include <hip/hip_bf16.h>
#include <math.h>

// ---------------- problem constants ----------------
#define B_     4
#define S_     512
#define SENC_  512
#define D_     1024
#define H_     16
#define FF_    4096
#define DH_    64
#define MROWS  (B_ * S_)          // 2048
#define NEGV   (-1e20f)
#define THRESH_ 0.99f

typedef __attribute__((ext_vector_type(16))) __bf16 v16bf;
typedef __attribute__((ext_vector_type(8)))  float  v8f;

// ---------------- reductions ----------------
__device__ __forceinline__ float blk_sum256(float v, float* sm) {
#pragma unroll
  for (int o = 16; o > 0; o >>= 1) v += __shfl_xor(v, o, 32);
  int lane = threadIdx.x & 31, w = threadIdx.x >> 5;
  __syncthreads();
  if (lane == 0) sm[w] = v;
  __syncthreads();
  float t = (threadIdx.x < 8) ? sm[threadIdx.x] : 0.f;
#pragma unroll
  for (int o = 4; o > 0; o >>= 1) t += __shfl_xor(t, o, 32);
  if (threadIdx.x == 0) sm[0] = t;
  __syncthreads();
  return sm[0];
}

// ---------------- tiny utility kernels ----------------
// cast + transpose: out[n*K + k] = (bf16) in[k*N + n]
__global__ void castT_bf16_k(const float* __restrict__ in, __bf16* __restrict__ out,
                             int K, int N) {
  size_t i = (size_t)blockIdx.x * blockDim.x + threadIdx.x;   // over K*N
  int k = (int)(i / N);
  int n = (int)(i - (size_t)k * N);
  out[(size_t)n * K + k] = (__bf16)in[i];
}

__global__ void zero_k(float* __restrict__ p, int n) {
  int i = blockIdx.x * blockDim.x + threadIdx.x;
  if (i < n) p[i] = 0.f;
}

__global__ void embed_k(const int* __restrict__ tokens, const float* __restrict__ emb,
                        float* __restrict__ x) {
  size_t i = (size_t)blockIdx.x * blockDim.x + threadIdx.x;   // over MROWS*D_
  size_t row = i >> 10;
  int d = (int)(i & (D_ - 1));
  x[i] = emb[(size_t)tokens[row] * D_ + d] * 32.0f;           // sqrt(1024)
}

__global__ void blend_k(float* __restrict__ prev, const float* __restrict__ x,
                        const float* __restrict__ uw) {
  size_t i = (size_t)blockIdx.x * blockDim.x + threadIdx.x;   // over MROWS*D_
  float u = uw[i >> 10];
  prev[i] = x[i] * u + prev[i] * (1.f - u);
}

// pos+time codes, ACT halting update; one block per (b,s) row
__global__ __launch_bounds__(256) void halt_k(float* __restrict__ x,
                                              const float* __restrict__ aw,
                                              const float* __restrict__ ab,
                                              float* __restrict__ hp,
                                              float* __restrict__ rem,
                                              float* __restrict__ uw, int t) {
  __shared__ float sm[8];
  int row = blockIdx.x;
  int spos = row & (S_ - 1);
  int tid = threadIdx.x;
  float loc = 0.f;
#pragma unroll
  for (int i = 0; i < 4; ++i) {
    int d = tid + i * 256;
    float inv = powf(10000.0f, -2.0f * (float)(d >> 1) / (float)D_);
    float ps = (d & 1) ? cosf((float)spos * inv) : sinf((float)spos * inv);
    float ts = (d & 1) ? cosf((float)t * inv)    : sinf((float)t * inv);
    float v = x[(size_t)row * D_ + d] + ps + ts;
    x[(size_t)row * D_ + d] = v;
    loc += v * aw[d];
  }
  float dot = blk_sum256(loc, sm);
  if (tid == 0) {
    float p = 1.f / (1.f + expf(-(dot + ab[0])));
    float h = hp[row], r = rem[row];
    float still = (h < 1.f) ? 1.f : 0.f;
    float cond = h + p * still;
    float nh  = (cond > THRESH_) ? still : 0.f;
    float st2 = (cond <= THRESH_) ? still : 0.f;
    h += p * st2;
    r += nh * (1.f - h);
    h += nh * r;
    hp[row] = h; rem[row] = r;
    uw[row] = p * st2 + nh * r;
  }
}

// x = LayerNorm(inp + x) * g + b ; one block per row
__global__ __launch_bounds__(256) void ln_res_k(const float* __restrict__ inp,
                                                float* __restrict__ x,
                                                const float* __restrict__ g,
                                                const float* __restrict__ bta) {
  __shared__ float sm[8];
  size_t row = blockIdx.x;
  int tid = threadIdx.x;
  float s[4];
  float loc = 0.f;
#pragma unroll
  for (int i = 0; i < 4; ++i) {
    int d = tid + i * 256;
    s[i] = inp[row * D_ + d] + x[row * D_ + d];
    loc += s[i];
  }
  float mean = blk_sum256(loc, sm) * (1.0f / D_);
  float lv = 0.f;
#pragma unroll
  for (int i = 0; i < 4; ++i) { float tt = s[i] - mean; lv += tt * tt; }
  float var = blk_sum256(lv, sm) * (1.0f / D_);
  float rstd = rsqrtf(var + 1e-12f);
#pragma unroll
  for (int i = 0; i < 4; ++i) {
    int d = tid + i * 256;
    x[row * D_ + d] = (s[i] - mean) * rstd * g[d] + bta[d];
  }
}

// ---------------- WMMA GEMM: C[M,N] = act(A[M,K] @ W[K,N] + bias) ----------------
// Wt is the transposed weight, [N][K] bf16, so every B fragment is two contiguous
// 16B LDS reads. block = 256 threads (8 waves); block tile 128x128; wave tile
// 32x64; K-step 32. B tile staged memory->LDS with GLOBAL_LOAD_ASYNC_TO_LDS_B128.
__global__ __launch_bounds__(256) void gemm_wmma_k(const float* __restrict__ A,
                                                   const __bf16* __restrict__ Wt,
                                                   const float* __restrict__ bias,
                                                   float* __restrict__ C,
                                                   int M, int N, int K, int relu) {
  __shared__ __attribute__((aligned(16))) __bf16 As[128][34];
  __shared__ __attribute__((aligned(16))) __bf16 Bst[128][40];  // [n][k], 80B pitch

  int tid = threadIdx.x, lane = tid & 31, wid = tid >> 5;
  int bm = blockIdx.y * 128, bn = blockIdx.x * 128;
  int mo = (wid & 3) * 32, no = (wid >> 2) * 64;
  int kb  = (lane < 16) ? 0 : 8;     // A fragment K base
  int kb2 = (lane < 16) ? 0 : 16;    // unused; B fragment also uses kb

  v8f vz = {0.f,0.f,0.f,0.f,0.f,0.f,0.f,0.f};
  v8f acc[2][4];
#pragma unroll
  for (int mt = 0; mt < 2; ++mt)
#pragma unroll
    for (int nt = 0; nt < 4; ++nt) acc[mt][nt] = vz;

  for (int k0 = 0; k0 < K; k0 += 32) {
    // stage B tile 128(n) x 32(k) bf16 memory -> LDS (async copy, ASYNCcnt)
#pragma unroll
    for (int p = 0; p < 2; ++p) {
      int sg = tid + p * 256;        // 512 x 16B segments
      int n  = sg >> 2;              // 0..127
      int c8 = (sg & 3) * 8;         // 0,8,16,24 (bf16)
      const __bf16* gptr = Wt + (size_t)(bn + n) * K + k0 + c8;
      unsigned int lds_off = (unsigned int)(size_t)(&Bst[n][c8]);
      asm volatile("global_load_async_to_lds_b128 %0, %1, off"
                   :: "v"(lds_off), "v"((unsigned long long)(size_t)gptr)
                   : "memory");
    }
    // stage A 128x32 (f32 -> bf16 through VGPRs); issue all loads first so they
    // overlap (staggered s_wait_loadcnt instead of serialized wait-0)
    {
      int rlo = tid >> 3;            // 0..31
      int c4  = (tid & 7) * 4;
      float4 fa[4];
#pragma unroll
      for (int p = 0; p < 4; ++p)
        fa[p] = *reinterpret_cast<const float4*>(A + (size_t)(bm + rlo + p * 32) * K + k0 + c4);
#pragma unroll
      for (int p = 0; p < 4; ++p) {
        int rr = rlo + p * 32;
        As[rr][c4 + 0] = (__bf16)fa[p].x; As[rr][c4 + 1] = (__bf16)fa[p].y;
        As[rr][c4 + 2] = (__bf16)fa[p].z; As[rr][c4 + 3] = (__bf16)fa[p].w;
      }
    }
    asm volatile("s_wait_asynccnt 0x0" ::: "memory");
    __syncthreads();

    if (k0 + 32 < K) {               // hint next tiles (global_prefetch_b8)
      __builtin_prefetch(A + (size_t)(bm + (tid >> 3)) * K + k0 + 32, 0, 0);
      __builtin_prefetch(Wt + (size_t)(bn + (tid >> 1)) * K + k0 + 32, 0, 0);
    }

    v16bf af[2];
#pragma unroll
    for (int mt = 0; mt < 2; ++mt) {
      int rr = mo + mt * 16 + (lane & 15);
#pragma unroll
      for (int e = 0; e < 8; ++e) af[mt][e]     = As[rr][kb + e];
#pragma unroll
      for (int e = 0; e < 8; ++e) af[mt][8 + e] = As[rr][16 + kb + e];
    }
#pragma unroll
    for (int nt = 0; nt < 4; ++nt) {
      v16bf bfg;
      int cc = no + nt * 16 + (lane & 15);
#pragma unroll
      for (int e = 0; e < 16; ++e) bfg[e] = Bst[cc][(e & 8) * 2 + kb + (e & 7)];   // two 16B runs (table order)
#pragma unroll
      for (int mt = 0; mt < 2; ++mt)
        acc[mt][nt] = __builtin_amdgcn_wmma_f32_16x16x32_bf16(
            false, af[mt], false, bfg, (short)0, acc[mt][nt], false, false);
    }
    __syncthreads();
  }

#pragma unroll
  for (int mt = 0; mt < 2; ++mt)
#pragma unroll
    for (int nt = 0; nt < 4; ++nt)
#pragma unroll
      for (int r = 0; r < 8; ++r) {
        int gm = bm + mo + mt * 16 + ((lane < 16) ? r : r + 8);
        int gn = bn + no + nt * 16 + (lane & 15);
        float v = acc[mt][nt][r] + bias[gn];
        if (relu) v = fmaxf(v, 0.f);
        C[(size_t)gm * N + gn] = v;
      }
}

// ---------------- WMMA flash attention ----------------
// grid: (S_/64, H_, B_), block = 128 threads (4 waves, 16 q rows each).
template <bool CAUSAL>
__global__ __launch_bounds__(128) void attn_k(const float* __restrict__ Q,
                                              const float* __restrict__ Kb,
                                              const float* __restrict__ Vb,
                                              const int* __restrict__ emask,
                                              float* __restrict__ O, int S_k) {
  __shared__ __bf16 Ks[32][66];      // [key][d]  (score B frag contiguous in d)
  __shared__ __bf16 Vst[64][40];     // [d][key]  (out   B frag contiguous in key)
  __shared__ __bf16 Ps[4][16][36];

  int tid = threadIdx.x, lane = tid & 31, wid = tid >> 5;
  int b = blockIdx.z, h = blockIdx.y;
  int qbase = blockIdx.x * 64;
  int kbA  = (lane < 16) ? 0 : 8;
  int kb2  = (lane < 16) ? 0 : 16;
  const float qscale = 0.125f;       // 1/sqrt(64)

  // stage Q fragments (16x64 per wave -> 2 A-fragments of K=32)
  v16bf qf[2];
  {
    int qrow = qbase + wid * 16 + (lane & 15);
    const float* qp = Q + ((size_t)(b * S_ + qrow)) * D_ + h * DH_;
#pragma unroll
    for (int kk = 0; kk < 2; ++kk) {
#pragma unroll
      for (int e = 0; e < 8; ++e)
        qf[kk][e]     = (__bf16)(qp[kk * 32 + kbA + e] * qscale);
#pragma unroll
      for (int e = 0; e < 8; ++e)
        qf[kk][8 + e] = (__bf16)(qp[kk * 32 + 16 + kbA + e] * qscale);
    }
  }

  v8f vz = {0.f,0.f,0.f,0.f,0.f,0.f,0.f,0.f};
  v8f oacc[4];
#pragma unroll
  for (int dt = 0; dt < 4; ++dt) oacc[dt] = vz;
  float mrow[8], lrow[8];
#pragma unroll
  for (int r = 0; r < 8; ++r) { mrow[r] = -1e30f; lrow[r] = 0.f; }

  int nchunks = CAUSAL ? (blockIdx.x * 2 + 2) : (S_k / 32);

  for (int kc = 0; kc < nchunks; ++kc) {
    // stage K (row-major) and V (transposed) chunk, f32 -> bf16
    for (int i = tid; i < 32 * 64; i += 128) {
      int kr = i >> 6, dc = i & 63;
      size_t gi = ((size_t)(b * S_k + kc * 32 + kr)) * D_ + h * DH_ + dc;
      Ks[kr][dc]  = (__bf16)Kb[gi];
      Vst[dc][kr] = (__bf16)Vb[gi];
    }
    __syncthreads();

    // scores: 2 tiles of 16 keys, contraction over DH=64 (2 WMMAs each)
    v8f sc[2];
#pragma unroll
    for (int nt = 0; nt < 2; ++nt) {
      v8f s = vz;
      int keyl = nt * 16 + (lane & 15);
#pragma unroll
      for (int kk = 0; kk < 2; ++kk) {
        v16bf bk;
#pragma unroll
        for (int e = 0; e < 16; ++e) bk[e] = Ks[keyl][kk * 32 + (e & 8) * 2 + kbA + (e & 7)];
        s = __builtin_amdgcn_wmma_f32_16x16x32_bf16(false, qf[kk], false, bk,
                                                    (short)0, s, false, false);
      }
      sc[nt] = s;
    }

    // masking
#pragma unroll
    for (int nt = 0; nt < 2; ++nt)
#pragma unroll
      for (int r = 0; r < 8; ++r) {
        int keyg = kc * 32 + nt * 16 + (lane & 15);
        int rowg = qbase + wid * 16 + ((lane < 16) ? r : r + 8);
        bool ok = CAUSAL ? (keyg <= rowg) : (emask[(size_t)b * S_k + keyg] != 0);
        if (!ok) sc[nt][r] = NEGV;
      }

    // online softmax (row = C-element index r within 16-lane half)
    float alpha[8];
#pragma unroll
    for (int r = 0; r < 8; ++r) {
      float mx = fmaxf(sc[0][r], sc[1][r]);
#pragma unroll
      for (int o = 8; o > 0; o >>= 1) mx = fmaxf(mx, __shfl_xor(mx, o, 32));
      float mnew = fmaxf(mrow[r], mx);
      float a  = expf(mrow[r] - mnew);
      float p0 = expf(sc[0][r] - mnew);
      float p1 = expf(sc[1][r] - mnew);
      float sum = p0 + p1;
#pragma unroll
      for (int o = 8; o > 0; o >>= 1) sum += __shfl_xor(sum, o, 32);
      lrow[r] = lrow[r] * a + sum;
      mrow[r] = mnew;
      alpha[r] = a;
      sc[0][r] = p0; sc[1][r] = p1;
    }
#pragma unroll
    for (int dt = 0; dt < 4; ++dt)
#pragma unroll
      for (int r = 0; r < 8; ++r) oacc[dt][r] *= alpha[r];

    // relayout P (C-layout -> A-layout) via per-wave LDS (in-order within wave)
#pragma unroll
    for (int r = 0; r < 8; ++r) {
      int prow = (lane < 16) ? r : r + 8;
      Ps[wid][prow][(lane & 15)]      = (__bf16)sc[0][r];
      Ps[wid][prow][16 + (lane & 15)] = (__bf16)sc[1][r];
    }
    v16bf pa;
    {
      int pr = lane & 15;
#pragma unroll
      for (int e = 0; e < 8; ++e) pa[e]     = Ps[wid][pr][kbA + e];
#pragma unroll
      for (int e = 0; e < 8; ++e) pa[8 + e] = Ps[wid][pr][16 + kbA + e];
    }

    // O += P @ V  (contraction over 32 keys; 4 d-tiles of 16)
#pragma unroll
    for (int dt = 0; dt < 4; ++dt) {
      v16bf bv;
      int dc = dt * 16 + (lane & 15);
#pragma unroll
      for (int e = 0; e < 16; ++e) bv[e] = Vst[dc][(e & 8) * 2 + kbA + (e & 7)];    // two 16B runs (table order)
      oacc[dt] = __builtin_amdgcn_wmma_f32_16x16x32_bf16(false, pa, false, bv,
                                                         (short)0, oacc[dt], false, false);
    }
    __syncthreads();
  }

  // normalize + write
#pragma unroll
  for (int dt = 0; dt < 4; ++dt)
#pragma unroll
    for (int r = 0; r < 8; ++r) {
      int rowg = qbase + wid * 16 + ((lane < 16) ? r : r + 8);
      int dc = h * DH_ + dt * 16 + (lane & 15);
      O[((size_t)(b * S_ + rowg)) * D_ + dc] = oacc[dt][r] / fmaxf(lrow[r], 1e-20f);
    }
}

// ---------------- host orchestration ----------------
static void gemm(const float* A, const __bf16* Wt, const float* bias, float* C,
                 int M, int N, int K, int relu, hipStream_t s) {
  dim3 g(N / 128, M / 128), b(256);
  gemm_wmma_k<<<g, b, 0, s>>>(A, Wt, bias, C, M, N, K, relu);
}

extern "C" void kernel_launch(void* const* d_in, const int* in_sizes, int n_in,
                              void* d_out, int out_size, void* d_ws, size_t ws_size,
                              hipStream_t stream) {
  (void)in_sizes; (void)n_in; (void)out_size; (void)ws_size;

  const int*   tokens   = (const int*)d_in[0];
  const float* enc_out  = (const float*)d_in[1];
  const int*   enc_mask = (const int*)d_in[2];
  const float* emb      = (const float*)d_in[3];
  const float* sa_qw = (const float*)d_in[4],  *sa_qb = (const float*)d_in[5];
  const float* sa_kw = (const float*)d_in[6],  *sa_kb = (const float*)d_in[7];
  const float* sa_vw = (const float*)d_in[8],  *sa_vb = (const float*)d_in[9];
  const float* sa_ow = (const float*)d_in[10], *sa_ob = (const float*)d_in[11];
  const float* ca_qw = (const float*)d_in[12], *ca_qb = (const float*)d_in[13];
  const float* ca_kw = (const float*)d_in[14], *ca_kb = (const float*)d_in[15];
  const float* ca_vw = (const float*)d_in[16], *ca_vb = (const float*)d_in[17];
  const float* ca_ow = (const float*)d_in[18], *ca_ob = (const float*)d_in[19];
  const float* ln1_g = (const float*)d_in[20], *ln1_b = (const float*)d_in[21];
  const float* ln2_g = (const float*)d_in[22], *ln2_b = (const float*)d_in[23];
  const float* ln3_g = (const float*)d_in[24], *ln3_b = (const float*)d_in[25];
  const float* ff_w1 = (const float*)d_in[26], *ff_b1 = (const float*)d_in[27];
  const float* ff_w2 = (const float*)d_in[28], *ff_b2 = (const float*)d_in[29];
  const float* act_w = (const float*)d_in[30], *act_b = (const float*)d_in[31];

  float* prev = (float*)d_out;                 // [B,S,D] output accumulator

  // ---- workspace layout ----
  size_t off = 0;
  auto alloc = [&](size_t bytes) {
    void* p = (char*)d_ws + off;
    off += (bytes + 255) & ~(size_t)255;
    return p;
  };
  const size_t MD = (size_t)MROWS * D_;
  float* x    = (float*)alloc(MD * 4);
  float* qb   = (float*)alloc(MD * 4);
  float* kbuf = (float*)alloc(MD * 4);
  float* vbuf = (float*)alloc(MD * 4);
  float* attn = (float*)alloc(MD * 4);
  float* proj = (float*)alloc(MD * 4);
  float* kenc = (float*)alloc(MD * 4);
  float* venc = (float*)alloc(MD * 4);
  float* ff1  = (float*)alloc((size_t)MROWS * FF_ * 4);
  float* hp   = (float*)alloc(MROWS * 4);
  float* rem  = (float*)alloc(MROWS * 4);
  float* uw   = (float*)alloc(MROWS * 4);
  __bf16* w_saq = (__bf16*)alloc((size_t)D_ * D_ * 2);
  __bf16* w_sak = (__bf16*)alloc((size_t)D_ * D_ * 2);
  __bf16* w_sav = (__bf16*)alloc((size_t)D_ * D_ * 2);
  __bf16* w_sao = (__bf16*)alloc((size_t)D_ * D_ * 2);
  __bf16* w_caq = (__bf16*)alloc((size_t)D_ * D_ * 2);
  __bf16* w_cak = (__bf16*)alloc((size_t)D_ * D_ * 2);
  __bf16* w_cav = (__bf16*)alloc((size_t)D_ * D_ * 2);
  __bf16* w_cao = (__bf16*)alloc((size_t)D_ * D_ * 2);
  __bf16* w_ff1 = (__bf16*)alloc((size_t)D_ * FF_ * 2);
  __bf16* w_ff2 = (__bf16*)alloc((size_t)FF_ * D_ * 2);

  // ---- weight casts (fp32 -> bf16, transposed to [N][K]), once per call ----
  auto castT = [&](const float* src, __bf16* dst, int K, int N) {
    size_t n = (size_t)K * N;
    castT_bf16_k<<<(unsigned)((n + 255) / 256), 256, 0, stream>>>(src, dst, K, N);
  };
  castT(sa_qw, w_saq, D_, D_);  castT(sa_kw, w_sak, D_, D_);
  castT(sa_vw, w_sav, D_, D_);  castT(sa_ow, w_sao, D_, D_);
  castT(ca_qw, w_caq, D_, D_);  castT(ca_kw, w_cak, D_, D_);
  castT(ca_vw, w_cav, D_, D_);  castT(ca_ow, w_cao, D_, D_);
  castT(ff_w1, w_ff1, D_, FF_); castT(ff_w2, w_ff2, FF_, D_);

  // ---- init state ----
  zero_k<<<(unsigned)((MD + 255) / 256), 256, 0, stream>>>(prev, (int)MD);
  zero_k<<<(MROWS + 255) / 256, 256, 0, stream>>>(hp, MROWS);
  zero_k<<<(MROWS + 255) / 256, 256, 0, stream>>>(rem, MROWS);

  // x = emb[tokens] * sqrt(D)
  embed_k<<<(unsigned)(MD / 256), 256, 0, stream>>>(tokens, emb, x);

  // hop-invariant encoder K/V projections (B*SENC == MROWS)
  gemm(enc_out, w_cak, ca_kb, kenc, MROWS, D_, D_, 0, stream);
  gemm(enc_out, w_cav, ca_vb, venc, MROWS, D_, D_, 0, stream);

  dim3 agrid(S_ / 64, H_, B_), ablk(128);

  for (int t = 0; t < 6; ++t) {
    // codes + ACT halting
    halt_k<<<MROWS, 256, 0, stream>>>(x, act_w, act_b, hp, rem, uw, t);

    // --- self attention ---
    gemm(x, w_saq, sa_qb, qb,   MROWS, D_, D_, 0, stream);
    gemm(x, w_sak, sa_kb, kbuf, MROWS, D_, D_, 0, stream);
    gemm(x, w_sav, sa_vb, vbuf, MROWS, D_, D_, 0, stream);
    attn_k<true><<<agrid, ablk, 0, stream>>>(qb, kbuf, vbuf, nullptr, attn, S_);
    gemm(attn, w_sao, sa_ob, proj, MROWS, D_, D_, 0, stream);
    ln_res_k<<<MROWS, 256, 0, stream>>>(proj, x, ln1_g, ln1_b);

    // --- cross attention ---
    gemm(x, w_caq, ca_qb, qb, MROWS, D_, D_, 0, stream);
    attn_k<false><<<agrid, ablk, 0, stream>>>(qb, kenc, venc, enc_mask, attn, SENC_);
    gemm(attn, w_cao, ca_ob, proj, MROWS, D_, D_, 0, stream);
    ln_res_k<<<MROWS, 256, 0, stream>>>(proj, x, ln2_g, ln2_b);

    // --- FFN ---
    gemm(x,   w_ff1, ff_b1, ff1,  MROWS, FF_, D_, 1, stream);
    gemm(ff1, w_ff2, ff_b2, proj, MROWS, D_, FF_, 0, stream);
    ln_res_k<<<MROWS, 256, 0, stream>>>(proj, x, ln3_g, ln3_b);

    // --- ACT blend ---
    blend_k<<<(unsigned)(MD / 256), 256, 0, stream>>>(prev, x, uw);
  }
}
